// GPT_30408368456372
// MI455X (gfx1250) — matrix operand order repaired and checked by reading, not compiled, not run on hardware
//
#include <hip/hip_runtime.h>
#include <hip/hip_bf16.h>
#include <math.h>

// ---------------------------------------------------------------------------
// GPT forward for MI455X (gfx1250): all GEMMs via V_WMMA_F32_16X16X32_F16.
// Weights are converted f32->f16 and transposed to [N,K] in workspace once per
// launch so every WMMA B-fragment is two 16B loads from one row (lane=col, K contig).
// Requires ws_size >= ~240 MB (activations + f16 transposed weights).
// ---------------------------------------------------------------------------

static constexpr int kB  = 2;
static constexpr int kS  = 1024;
static constexpr int kD  = 1024;
static constexpr int kH  = 16;
static constexpr int kL  = 4;
static constexpr int kF  = 4096;
static constexpr int kV  = 32000;
static constexpr int kHD = 64;
static constexpr int kM  = kB * kS;   // 2048 token rows

typedef _Float16 v16h __attribute__((ext_vector_type(16)));
typedef _Float16 v8h  __attribute__((ext_vector_type(8)));
typedef float    v8f  __attribute__((ext_vector_type(8)));

// D = A(16x32 f16) * B(32x16 f16) + C(16x16 f32)
__device__ inline v8f wmma32(v16h a, v16h b, v8f c) {
  return __builtin_amdgcn_wmma_f32_16x16x32_f16(
      /*neg_a=*/false, a, /*neg_b=*/false, b,
      /*c_mod=*/(short)0, c, /*reuse_a=*/false, /*reuse_b=*/false);
}

// A-fragment style: 8 halves at p, 8 halves at p+16 (ISA 16-bit A 16x32 layout)
__device__ inline v16h load16_gap(const _Float16* p) {
  v8h lo = *(const v8h*)p;
  v8h hi = *(const v8h*)(p + 16);
  v16h r;
#pragma unroll
  for (int i = 0; i < 8; ++i) { r[i] = lo[i]; r[i + 8] = hi[i]; }
  return r;
}
// B-fragment style: 8 halves at p, 8 halves at p+16 (same K order as A)
__device__ inline v16h load16(const _Float16* p) { return load16_gap(p); }

// ---------------------------------------------------------------------------
// Embedding + sinusoidal positional encoding -> x (f32) and x_h (f16)
// ---------------------------------------------------------------------------
__global__ __launch_bounds__(256)
void embed_kernel(const int* __restrict__ tokens, const float* __restrict__ emb,
                  float* __restrict__ xf, _Float16* __restrict__ xh) {
  size_t idx = (size_t)blockIdx.x * 256 + threadIdx.x;   // over kM*kD
  int d = (int)(idx % kD);
  size_t bs = idx / kD;
  int s = (int)(bs % kS);
  int tok = tokens[bs];
  float e = emb[(size_t)tok * kD + d];
  float freq = expf((float)(d & ~1) * (-9.210340371976184f / (float)kD)); // ln(1e4)
  float ang = (float)s * freq;
  float pe = (d & 1) ? cosf(ang) : sinf(ang);
  float x = e + pe;
  xf[idx] = x;
  xh[idx] = (_Float16)x;
}

// ---------------------------------------------------------------------------
// f32 [R,C] -> f16 [C,R]  (weight conversion + transpose), R,C multiples of 32
// ---------------------------------------------------------------------------
__global__ __launch_bounds__(256)
void transpose_cvt_f32(const float* __restrict__ W, _Float16* __restrict__ Wt,
                       int R, int C) {
  __shared__ float tile[32][33];
  int c0 = blockIdx.x * 32, r0 = blockIdx.y * 32;
  int tx = threadIdx.x, ty = threadIdx.y;          // block (32,8)
#pragma unroll
  for (int i = 0; i < 4; ++i)
    tile[ty + 8 * i][tx] = W[(size_t)(r0 + ty + 8 * i) * C + (c0 + tx)];
  __syncthreads();
#pragma unroll
  for (int i = 0; i < 4; ++i)
    Wt[(size_t)(c0 + ty + 8 * i) * R + (r0 + tx)] = (_Float16)tile[tx][ty + 8 * i];
}

// f16 [R,C] -> f16 [C,R], batched over blockIdx.z (used for V -> V^T per batch)
__global__ __launch_bounds__(256)
void transpose_f16_batched(const _Float16* __restrict__ in,
                           _Float16* __restrict__ out, int R, int C) {
  __shared__ _Float16 tile[32][34];
  size_t base = (size_t)blockIdx.z * R * C;
  int c0 = blockIdx.x * 32, r0 = blockIdx.y * 32;
  int tx = threadIdx.x, ty = threadIdx.y;          // block (32,8)
#pragma unroll
  for (int i = 0; i < 4; ++i)
    tile[ty + 8 * i][tx] = in[base + (size_t)(r0 + ty + 8 * i) * C + (c0 + tx)];
  __syncthreads();
#pragma unroll
  for (int i = 0; i < 4; ++i)
    out[base + (size_t)(c0 + ty + 8 * i) * R + (r0 + tx)] = tile[tx][ty + 8 * i];
}

// ---------------------------------------------------------------------------
// WMMA GEMM:  C[M,N] = A[M,K](f16) * Bt[N,K]^T(f16) + bias, optional exact GELU
// 4 waves/block, each wave computes a 32x64 tile (8 accumulators).
// M mult of 128, N mult of 64, K mult of 32.
// ---------------------------------------------------------------------------
__global__ __launch_bounds__(128)
void gemm_f16_kernel(const _Float16* __restrict__ A, const _Float16* __restrict__ Bt,
                     const float* __restrict__ bias, float* __restrict__ Cf,
                     _Float16* __restrict__ Ch, int M, int N, int K, int act_gelu) {
  const int lane  = threadIdx.x & 31;
  const int wave  = threadIdx.x >> 5;
  const int lm    = lane & 15;
  const int hi8   = (lane >> 4) * 8;     // C-layout row offset
  const int koffA = hi8;                 // A frag K base: 0 / 8
  const int koffB = hi8;                 // B frag K base: 0 / 8
  const int n0 = blockIdx.x * 64;
  const int m0 = (blockIdx.y * 4 + wave) * 32;

  v8f acc[2][4] = {};

  for (int k0 = 0; k0 < K; k0 += 32) {
    v16h a[2];
#pragma unroll
    for (int ms = 0; ms < 2; ++ms)
      a[ms] = load16_gap(A + (size_t)(m0 + ms * 16 + lm) * K + k0 + koffA);
#pragma unroll
    for (int ns = 0; ns < 4; ++ns) {
      v16h bfr = load16(Bt + (size_t)(n0 + ns * 16 + lm) * K + k0 + koffB);
      acc[0][ns] = wmma32(a[0], bfr, acc[0][ns]);
      acc[1][ns] = wmma32(a[1], bfr, acc[1][ns]);
    }
  }

#pragma unroll
  for (int ms = 0; ms < 2; ++ms)
#pragma unroll
    for (int ns = 0; ns < 4; ++ns)
#pragma unroll
      for (int r = 0; r < 8; ++r) {
        int row = m0 + ms * 16 + r + hi8;
        int col = n0 + ns * 16 + lm;
        float v = acc[ms][ns][r];
        if (bias) v += bias[col];
        if (act_gelu) v = 0.5f * v * (1.0f + erff(v * 0.70710678118654752f));
        if (Cf) Cf[(size_t)row * N + col] = v;
        if (Ch) Ch[(size_t)row * N + col] = (_Float16)v;
      }
}

// ---------------------------------------------------------------------------
// Flash-style causal attention. One wave per (b, h, 16-query tile).
// qh/kh: [B*S, D] f16 (head h at cols h*64..), vT: [B*H*64, S] f16.
// Writes oh: [B*S, D] f16 (heads concatenated).
// ---------------------------------------------------------------------------
__global__ __launch_bounds__(128)
void attn_kernel(const _Float16* __restrict__ qh, const _Float16* __restrict__ kh,
                 const _Float16* __restrict__ vT, _Float16* __restrict__ oh) {
  __shared__ __align__(32) _Float16 pbuf[4][16 * 32];
  const int lane  = threadIdx.x & 31;
  const int wave  = threadIdx.x >> 5;
  const int lm    = lane & 15;
  const int hi8   = (lane >> 4) * 8;
  const int koffA = hi8;
  const int koffB = hi8;
  const int qt = blockIdx.x * 4 + wave;      // query tile 0..63
  const int bh = blockIdx.y;                 // 0..B*H-1
  const int b = bh / kH, h = bh % kH;

  // Q A-fragments for K-dim 0..63 (two K=32 chunks)
  const _Float16* qp = qh + ((size_t)(b * kS + qt * 16 + lm)) * kD + h * kHD;
  v16h qa[2];
  qa[0] = load16_gap(qp + koffA);
  qa[1] = load16_gap(qp + 32 + koffA);

  float mrow[8], lrow[8];
  v8f o[4] = {};
#pragma unroll
  for (int r = 0; r < 8; ++r) { mrow[r] = -INFINITY; lrow[r] = 0.0f; }

  _Float16* pb = pbuf[wave];
  const int nch = qt / 2 + 1;                // 32-key chunks covering causal span
  for (int c = 0; c < nch; ++c) {
    const int c0 = c * 32;
    v8f s[2] = {};
#pragma unroll
    for (int ct = 0; ct < 2; ++ct) {         // scores [16, 32] = Q * K^T
      const _Float16* kp =
          kh + ((size_t)(b * kS + c0 + ct * 16 + lm)) * kD + h * kHD + koffB;
      v16h kf0 = load16(kp);
      v16h kf1 = load16(kp + 32);
      s[ct] = wmma32(qa[0], kf0, s[ct]);
      s[ct] = wmma32(qa[1], kf1, s[ct]);
    }
    // scale + causal mask
#pragma unroll
    for (int ct = 0; ct < 2; ++ct)
#pragma unroll
      for (int r = 0; r < 8; ++r) {
        int col  = c0 + ct * 16 + lm;
        int rowq = qt * 16 + r + hi8;
        float val = s[ct][r] * 0.125f;       // 1/sqrt(64)
        if (col > rowq) val = -INFINITY;
        s[ct][r] = val;
      }
    // per-row max across 16 lanes (xor 1..8 keeps lane halves independent)
    float chmax[8];
#pragma unroll
    for (int r = 0; r < 8; ++r) chmax[r] = fmaxf(s[0][r], s[1][r]);
#pragma unroll
    for (int off = 1; off < 16; off <<= 1)
#pragma unroll
      for (int r = 0; r < 8; ++r)
        chmax[r] = fmaxf(chmax[r], __shfl_xor(chmax[r], off, 32));

    float rsum[8];
#pragma unroll
    for (int r = 0; r < 8; ++r) {
      float nm   = fmaxf(mrow[r], chmax[r]);
      float corr = __expf(mrow[r] - nm);
      float p0 = __expf(s[0][r] - nm);
      float p1 = __expf(s[1][r] - nm);
      s[0][r] = p0; s[1][r] = p1;
      rsum[r] = p0 + p1;
      lrow[r] *= corr;
      mrow[r] = nm;
#pragma unroll
      for (int ns = 0; ns < 4; ++ns) o[ns][r] *= corr;
    }
#pragma unroll
    for (int off = 1; off < 16; off <<= 1)
#pragma unroll
      for (int r = 0; r < 8; ++r) rsum[r] += __shfl_xor(rsum[r], off, 32);
#pragma unroll
    for (int r = 0; r < 8; ++r) lrow[r] += rsum[r];

    // stage P (C-layout) -> LDS -> reload as A-fragment (per-wave region)
#pragma unroll
    for (int ct = 0; ct < 2; ++ct)
#pragma unroll
      for (int r = 0; r < 8; ++r)
        pb[(r + hi8) * 32 + ct * 16 + lm] = (_Float16)s[ct][r];
    asm volatile("s_wait_dscnt 0" ::: "memory");
    v16h pa = load16_gap(pb + lm * 32 + koffA);

    // O += P[16,32] * V[32,64]
#pragma unroll
    for (int ns = 0; ns < 4; ++ns) {
      const _Float16* vp =
          vT + ((size_t)(bh * kHD + ns * 16 + lm)) * kS + c0 + koffB;
      v16h vf = load16(vp);
      o[ns] = wmma32(pa, vf, o[ns]);
    }
  }

#pragma unroll
  for (int r = 0; r < 8; ++r) {
    float inv = 1.0f / lrow[r];
    int row = b * kS + qt * 16 + r + hi8;
#pragma unroll
    for (int ns = 0; ns < 4; ++ns)
      oh[(size_t)row * kD + h * kHD + ns * 16 + lm] = (_Float16)(o[ns][r] * inv);
  }
}

// ---------------------------------------------------------------------------
// y = LayerNorm(a + b) * g + beta ; writes f32 and f16 copies. One block/row.
// ---------------------------------------------------------------------------
__global__ __launch_bounds__(256)
void add_ln_kernel(const float* __restrict__ A, const float* __restrict__ Bres,
                   const float* __restrict__ g, const float* __restrict__ beta,
                   float* __restrict__ Y, _Float16* __restrict__ Yh) {
  __shared__ float sbuf[256], sbuf2[256];
  int row = blockIdx.x;
  int tid = threadIdx.x;
  const float* a = A + (size_t)row * kD;
  const float* b = Bres + (size_t)row * kD;
  float v[4]; float s = 0.f, ss = 0.f;
#pragma unroll
  for (int i = 0; i < 4; ++i) {
    int d = tid + 256 * i;
    float t = a[d] + b[d];
    v[i] = t; s += t; ss += t * t;
  }
  sbuf[tid] = s; sbuf2[tid] = ss;
  __syncthreads();
  for (int st = 128; st > 0; st >>= 1) {
    if (tid < st) { sbuf[tid] += sbuf[tid + st]; sbuf2[tid] += sbuf2[tid + st]; }
    __syncthreads();
  }
  float mean = sbuf[0] * (1.0f / kD);
  float var  = sbuf2[0] * (1.0f / kD) - mean * mean;
  float rstd = rsqrtf(var + 1e-5f);
#pragma unroll
  for (int i = 0; i < 4; ++i) {
    int d = tid + 256 * i;
    float t = (v[i] - mean) * rstd * g[d] + beta[d];
    Y[(size_t)row * kD + d] = t;
    Yh[(size_t)row * kD + d] = (_Float16)t;
  }
}

// ---------------------------------------------------------------------------
extern "C" void kernel_launch(void* const* d_in, const int* in_sizes, int n_in,
                              void* d_out, int out_size, void* d_ws, size_t ws_size,
                              hipStream_t stream) {
  (void)in_sizes; (void)n_in; (void)out_size; (void)ws_size;
  const int*   tokens = (const int*)d_in[0];
  const float* emb  = (const float*)d_in[1];
  const float* Wq   = (const float*)d_in[2];
  const float* bq   = (const float*)d_in[3];
  const float* Wk   = (const float*)d_in[4];
  const float* bk   = (const float*)d_in[5];
  const float* Wv   = (const float*)d_in[6];
  const float* bv   = (const float*)d_in[7];
  const float* Wo   = (const float*)d_in[8];
  const float* bo   = (const float*)d_in[9];
  const float* ln1g = (const float*)d_in[10];
  const float* ln1b = (const float*)d_in[11];
  const float* W1   = (const float*)d_in[12];
  const float* b1   = (const float*)d_in[13];
  const float* W2   = (const float*)d_in[14];
  const float* b2   = (const float*)d_in[15];
  const float* ln2g = (const float*)d_in[16];
  const float* ln2b = (const float*)d_in[17];
  const float* Wout = (const float*)d_in[18];
  const float* bout = (const float*)d_in[19];
  float* logits = (float*)d_out;

  char* wsp = (char*)d_ws;
  size_t off = 0;
  auto alloc = [&](size_t bytes) -> void* {
    void* p = wsp + off;
    off += (bytes + 255) & ~(size_t)255;
    return p;
  };

  _Float16* xh   = (_Float16*)alloc((size_t)kM * kD * 2);
  float*    xf   = (float*)   alloc((size_t)kM * kD * 4);
  _Float16* qhb  = (_Float16*)alloc((size_t)kM * kD * 2);
  _Float16* khb  = (_Float16*)alloc((size_t)kM * kD * 2);
  _Float16* vhb  = (_Float16*)alloc((size_t)kM * kD * 2);
  _Float16* vTb  = (_Float16*)alloc((size_t)kM * kD * 2);
  _Float16* ohb  = (_Float16*)alloc((size_t)kM * kD * 2);
  float*    tmpf = (float*)   alloc((size_t)kM * kD * 4);  // att-out / ffn2-out
  float*    adnf = (float*)   alloc((size_t)kM * kD * 4);
  _Float16* adnh = (_Float16*)alloc((size_t)kM * kD * 2);
  _Float16* h1h  = (_Float16*)alloc((size_t)kM * kF * 2);
  _Float16* Wqt  = (_Float16*)alloc((size_t)kL * kD * kD * 2);
  _Float16* Wkt  = (_Float16*)alloc((size_t)kL * kD * kD * 2);
  _Float16* Wvt  = (_Float16*)alloc((size_t)kL * kD * kD * 2);
  _Float16* Wot  = (_Float16*)alloc((size_t)kL * kD * kD * 2);
  _Float16* W1t  = (_Float16*)alloc((size_t)kL * kD * kF * 2);
  _Float16* W2t  = (_Float16*)alloc((size_t)kL * kF * kD * 2);
  _Float16* Woutt= (_Float16*)alloc((size_t)kV * kD * 2);

  dim3 tb(32, 8);
  // Weight convert + transpose to [N,K] f16
  for (int l = 0; l < kL; ++l) {
    size_t wo = (size_t)l * kD * kD;
    transpose_cvt_f32<<<dim3(kD/32, kD/32), tb, 0, stream>>>(Wq + wo, Wqt + wo, kD, kD);
    transpose_cvt_f32<<<dim3(kD/32, kD/32), tb, 0, stream>>>(Wk + wo, Wkt + wo, kD, kD);
    transpose_cvt_f32<<<dim3(kD/32, kD/32), tb, 0, stream>>>(Wv + wo, Wvt + wo, kD, kD);
    transpose_cvt_f32<<<dim3(kD/32, kD/32), tb, 0, stream>>>(Wo + wo, Wot + wo, kD, kD);
    size_t fo = (size_t)l * kD * kF;
    transpose_cvt_f32<<<dim3(kF/32, kD/32), tb, 0, stream>>>(W1 + fo, W1t + fo, kD, kF);
    transpose_cvt_f32<<<dim3(kD/32, kF/32), tb, 0, stream>>>(W2 + fo, W2t + fo, kF, kD);
  }
  transpose_cvt_f32<<<dim3(kV/32, kD/32), tb, 0, stream>>>(Wout, Woutt, kD, kV);

  // Embedding + positional encoding
  embed_kernel<<<(kM * kD) / 256, 256, 0, stream>>>(tokens, emb, xf, xh);

  dim3 gD(kD / 64, kM / 128);   // GEMM grids (N/64, M/128)
  dim3 gF(kF / 64, kM / 128);
  dim3 gV(kV / 64, kM / 128);

  for (int l = 0; l < kL; ++l) {
    size_t wo = (size_t)l * kD * kD;
    size_t fo = (size_t)l * kD * kF;
    // QKV projections (f16 outputs only)
    gemm_f16_kernel<<<gD, 128, 0, stream>>>(xh, Wqt + wo, bq + l*kD, nullptr, qhb, kM, kD, kD, 0);
    gemm_f16_kernel<<<gD, 128, 0, stream>>>(xh, Wkt + wo, bk + l*kD, nullptr, khb, kM, kD, kD, 0);
    gemm_f16_kernel<<<gD, 128, 0, stream>>>(xh, Wvt + wo, bv + l*kD, nullptr, vhb, kM, kD, kD, 0);
    // V^T per batch: [S,D] -> [D,S]  (== [B*H*64, S] head-major)
    transpose_f16_batched<<<dim3(kD/32, kS/32, kB), tb, 0, stream>>>(vhb, vTb, kS, kD);
    // Causal flash attention
    attn_kernel<<<dim3(kS/64, kB*kH), 128, 0, stream>>>(qhb, khb, vTb, ohb);
    // Output projection (f32)
    gemm_f16_kernel<<<gD, 128, 0, stream>>>(ohb, Wot + wo, bo + l*kD, tmpf, nullptr, kM, kD, kD, 0);
    // adn = LN(x + att)
    add_ln_kernel<<<kM, 256, 0, stream>>>(xf, tmpf, ln1g + l*kD, ln1b + l*kD, adnf, adnh);
    // FFN: gelu(adn @ W1 + b1) @ W2 + b2
    gemm_f16_kernel<<<gF, 128, 0, stream>>>(adnh, W1t + fo, b1 + l*kF, nullptr, h1h, kM, kF, kD, 1);
    gemm_f16_kernel<<<gD, 128, 0, stream>>>(h1h, W2t + fo, b2 + l*kD, tmpf, nullptr, kM, kD, kF, 0);
    // x = LN(h + adn)
    add_ln_kernel<<<kM, 256, 0, stream>>>(tmpf, adnf, ln2g + l*kD, ln2b + l*kD, xf, xh);
  }

  // Final logits: [2048,1024] x [1024,32000] + bout -> f32 d_out
  gemm_f16_kernel<<<gV, 128, 0, stream>>>(xh, Woutt, bout, logits, nullptr, kM, kV, kD, 0);
}
